// ViTSelfAttention_8117488190189
// MI455X (gfx1250) — 8-bit matrix operand order repaired and checked by reading, compiled, not run on hardware
//
#include <hip/hip_runtime.h>
#include <hip/hip_bf16.h>
#include <stdint.h>


#define BB   64
#define SS   197
#define HH   768
#define NHH  12
#define HDD  64
#define MTOK (BB * SS)
#define MT16 (MTOK / 16)

typedef __attribute__((ext_vector_type(8))) int      v8i;
typedef __attribute__((ext_vector_type(4))) int      i32x4;
typedef unsigned long long u64;

union FragA { v8i v; u64 q[4]; };
union FragB { v8i v; i32x4 x[2]; };
union VRow  { i32x4 x[4]; signed char c[64]; };

template <bool SA, bool SB>
static __device__ __forceinline__ v8i wmma_iu8(v8i a, v8i b, v8i c) {
    return __builtin_amdgcn_wmma_i32_16x16x64_iu8(SA, a, SB, b, c, false, false);
}

__global__ __launch_bounds__(256)
void prep_w_kernel(const float* __restrict__ w, signed char* __restrict__ wb,
                   float* __restrict__ scl)
{
    __shared__ float red[256];
    const int n   = blockIdx.x;
    const int tid = threadIdx.x;
    const float* wr = w + (size_t)n * HH;
    float s = 0.f;
    for (int j = tid; j < HH; j += 256) s += fabsf(wr[j]);
    red[tid] = s;
    __syncthreads();
    for (int off = 128; off > 0; off >>= 1) {
        if (tid < off) red[tid] += red[tid + off];
        __syncthreads();
    }
    if (tid == 0) scl[n] = red[0] * (1.0f / HH);
    for (int j = tid; j < HH; j += 256)
        wb[(size_t)n * HH + j] = (wr[j] >= 0.f) ? (signed char)1 : (signed char)-1;
}

__global__ __launch_bounds__(256)
void binarize_x_kernel(const float* __restrict__ x,
                       const float* __restrict__ mq, const float* __restrict__ mk,
                       const float* __restrict__ mv,
                       signed char* __restrict__ xq, signed char* __restrict__ xk,
                       signed char* __restrict__ xv, int total)
{
    int i = blockIdx.x * 256 + threadIdx.x;
    if (i >= total) return;
    const int c = i % HH;
    const float v = x[i];
    xq[i] = ((v + mq[c]) >= 0.f) ? (signed char)1 : (signed char)-1;
    xk[i] = ((v + mk[c]) >= 0.f) ? (signed char)1 : (signed char)-1;
    xv[i] = ((v + mv[c]) >= 0.f) ? (signed char)1 : (signed char)-1;
}

__global__ __launch_bounds__(256)
void gemm_ln_kernel(const signed char* __restrict__ xb,
                    const signed char* __restrict__ wb,
                    const float* __restrict__ scl,
                    const float* __restrict__ bias,
                    const float* __restrict__ lng,
                    const float* __restrict__ lnb,
                    const float* __restrict__ xres,
                    const float* __restrict__ m1,
                    const float* __restrict__ aslope,
                    const float* __restrict__ m2,
                    const float* __restrict__ move2,
                    float* __restrict__ msum,
                    signed char* __restrict__ outb,
                    int accum)
{
    __shared__ float s_y[16 * HH];
    __shared__ float s_r1[256];
    __shared__ float s_r2[256];

    const int tid   = threadIdx.x;
    const int wave  = tid >> 5;
    const int lane  = tid & 31;
    const int half  = lane >> 4;
    const int l16   = lane & 15;
    const int mbase = blockIdx.x * 16;

    v8i acc[6];
    #pragma unroll
    for (int t = 0; t < 6; ++t) acc[t] = (v8i){0,0,0,0,0,0,0,0};

    const signed char* arow = xb + (size_t)(mbase + l16) * HH + half * 16;

    for (int kk = 0; kk < 12; ++kk) {
        const int k0 = kk * 64;
        const signed char* pa = arow + k0;
        FragA fa;
        fa.q[0] = *(const u64*)(pa);
        fa.q[1] = *(const u64*)(pa + 8);
        fa.q[2] = *(const u64*)(pa + 32);
        fa.q[3] = *(const u64*)(pa + 40);
        if (kk + 1 < 12) __builtin_prefetch(pa + 64, 0, 1);
        #pragma unroll
        for (int t = 0; t < 6; ++t) {
            const int n = (wave * 6 + t) * 16 + l16;
            const signed char* pb = wb + (size_t)n * HH + k0 + half * 16;
            FragB fb;
            fb.x[0] = *(const i32x4*)(pb);
            fb.x[1] = *(const i32x4*)(pb + 32);
            acc[t] = wmma_iu8<true, true>(fa.v, fb.v, acc[t]);
        }
    }

    #pragma unroll
    for (int t = 0; t < 6; ++t) {
        const int n  = (wave * 6 + t) * 16 + l16;
        const float sc = scl[n];
        const float bs = bias[n];
        #pragma unroll
        for (int r = 0; r < 8; ++r) {
            const int m = r + half * 8;
            s_y[m * HH + n] = (float)acc[t][r] * sc + bs;
        }
    }
    __syncthreads();

    const int row = tid >> 4;
    const int sub = tid & 15;
    float s1 = 0.f, s2 = 0.f;
    for (int j = sub; j < HH; j += 16) {
        const float v = s_y[row * HH + j];
        s1 += v; s2 += v * v;
    }
    s_r1[tid] = s1; s_r2[tid] = s2;
    __syncthreads();
    for (int off = 8; off > 0; off >>= 1) {
        if (sub < off) {
            s_r1[tid] += s_r1[tid + off];
            s_r2[tid] += s_r2[tid + off];
        }
        __syncthreads();
    }
    const float mu   = s_r1[row * 16] * (1.0f / HH);
    const float var  = s_r2[row * 16] * (1.0f / HH) - mu * mu;
    const float rstd = rsqrtf(var + 1e-12f);
    const int grow   = mbase + row;

    for (int j = sub; j < HH; j += 16) {
        const size_t gi = (size_t)grow * HH + j;
        const float ln  = (s_y[row * HH + j] - mu) * rstd * lng[j] + lnb[j];
        const float y2  = ln + xres[gi];
        const float tt  = y2 - m1[j];
        const float m   = (tt >= 0.f ? tt : aslope[j] * tt) + m2[j];
        msum[gi] = accum ? (msum[gi] + m) : m;
        outb[gi] = ((m + move2[j]) >= 0.f) ? (signed char)1 : (signed char)-1;
    }
}

__global__ __launch_bounds__(32)
void attn_kernel(const signed char* __restrict__ qb,
                 const signed char* __restrict__ kb,
                 const signed char* __restrict__ vb,
                 const float* __restrict__ clip,
                 float* __restrict__ ctx)
{
    __shared__ float s_sc[16 * 208];
    __shared__ __align__(16) signed char s_pb[16 * 256];
    __shared__ __align__(16) signed char s_vt[64 * 256];

    const int lane = threadIdx.x;
    const int half = lane >> 4;
    const int l16  = lane & 15;
    const int qt   = blockIdx.x;
    const int h    = blockIdx.y;
    const int b    = blockIdx.z;
    const float alpha = clip[0];

    int qrow = qt * 16 + l16; if (qrow > SS - 1) qrow = SS - 1;
    const signed char* pq = qb + ((size_t)(b * SS + qrow)) * HH + h * HDD + half * 16;
    FragA fq;
    fq.q[0] = *(const u64*)(pq);
    fq.q[1] = *(const u64*)(pq + 8);
    fq.q[2] = *(const u64*)(pq + 32);
    fq.q[3] = *(const u64*)(pq + 40);

    for (int j = lane; j < 256; j += 32) {
        const int jj = (j < SS) ? j : SS - 1;
        const signed char* pv = vb + ((size_t)(b * SS + jj)) * HH + h * HDD;
        VRow vr;
        vr.x[0] = *(const i32x4*)(pv);
        vr.x[1] = *(const i32x4*)(pv + 16);
        vr.x[2] = *(const i32x4*)(pv + 32);
        vr.x[3] = *(const i32x4*)(pv + 48);
        #pragma unroll
        for (int d = 0; d < 64; ++d) s_vt[d * 256 + j] = vr.c[d];
    }

    for (int kt = 0; kt < 13; ++kt) {
        int krow = kt * 16 + l16; if (krow > SS - 1) krow = SS - 1;
        const signed char* pk = kb + ((size_t)(b * SS + krow)) * HH + h * HDD + half * 16;
        FragB fk;
        fk.x[0] = *(const i32x4*)(pk);
        fk.x[1] = *(const i32x4*)(pk + 32);
        v8i sacc = (v8i){0,0,0,0,0,0,0,0};
        sacc = wmma_iu8<true, true>(fq.v, fk.v, sacc);
        #pragma unroll
        for (int r = 0; r < 8; ++r) {
            const int m = r + half * 8;
            s_sc[m * 208 + kt * 16 + l16] = (float)sacc[r] * 0.125f;
        }
    }
    __syncthreads();

    if (lane < 16) {
        const int m = lane;
        float mx = -1e30f;
        for (int j = 0; j < SS; ++j) mx = fmaxf(mx, s_sc[m * 208 + j]);
        float sm = 0.f;
        for (int j = 0; j < SS; ++j) sm += __expf(s_sc[m * 208 + j] - mx);
        const float th = 0.5f * alpha * sm;
        for (int j = 0; j < 256; ++j) {
            signed char p = 0;
            if (j < SS && __expf(s_sc[m * 208 + j] - mx) >= th) p = 1;
            s_pb[m * 256 + j] = p;
        }
    }
    __syncthreads();

    #pragma unroll
    for (int nt = 0; nt < 4; ++nt) {
        v8i cacc = (v8i){0,0,0,0,0,0,0,0};
        #pragma unroll
        for (int kc = 0; kc < 4; ++kc) {
            const signed char* pp = s_pb + l16 * 256 + kc * 64 + half * 16;
            FragA fp;
            fp.q[0] = *(const u64*)(pp);
            fp.q[1] = *(const u64*)(pp + 8);
            fp.q[2] = *(const u64*)(pp + 32);
            fp.q[3] = *(const u64*)(pp + 40);
            const signed char* pvt = s_vt + (nt * 16 + l16) * 256 + kc * 64 + half * 16;
            FragB fv;
            fv.x[0] = *(const i32x4*)(pvt);
            fv.x[1] = *(const i32x4*)(pvt + 32);
            cacc = wmma_iu8<false, true>(fp.v, fv.v, cacc);
        }
        #pragma unroll
        for (int r = 0; r < 8; ++r) {
            const int m   = r + half * 8;
            const int row = qt * 16 + m;
            if (row < SS)
                ctx[((size_t)(b * SS + row)) * HH + h * HDD + nt * 16 + l16] =
                    alpha * (float)cacc[r];
        }
    }
}

__global__ __launch_bounds__(256)
void final_ln_kernel(const float* __restrict__ ctx, const float* __restrict__ msum,
                     const float* __restrict__ g, const float* __restrict__ beta,
                     const float* __restrict__ m1, const float* __restrict__ aslope,
                     const float* __restrict__ m2, float* __restrict__ out)
{
    __shared__ float r1[256], r2[256];
    const int row = blockIdx.x;
    const int tid = threadIdx.x;
    const float* c = ctx + (size_t)row * HH;
    float v[3], s1 = 0.f, s2 = 0.f;
    #pragma unroll
    for (int i = 0; i < 3; ++i) {
        v[i] = c[tid + i * 256];
        s1 += v[i]; s2 += v[i] * v[i];
    }
    r1[tid] = s1; r2[tid] = s2;
    __syncthreads();
    for (int off = 128; off > 0; off >>= 1) {
        if (tid < off) { r1[tid] += r1[tid + off]; r2[tid] += r2[tid + off]; }
        __syncthreads();
    }
    const float mu   = r1[0] * (1.0f / HH);
    const float var  = r2[0] * (1.0f / HH) - mu * mu;
    const float rstd = rsqrtf(var + 1e-12f);
    #pragma unroll
    for (int i = 0; i < 3; ++i) {
        const int j = tid + i * 256;
        const size_t gi = (size_t)row * HH + j;
        const float ln = (v[i] - mu) * rstd * g[j] + beta[j];
        const float y2 = ln + msum[gi];
        const float t  = y2 - m1[j];
        out[gi] = (t >= 0.f ? t : aslope[j] * t) + m2[j];
    }
}

extern "C" void kernel_launch(void* const* d_in, const int* in_sizes, int n_in,
                              void* d_out, int out_size, void* d_ws, size_t ws_size,
                              hipStream_t stream)
{
    const float* x      = (const float*)d_in[0];
    const float* moveq  = (const float*)d_in[1];
    const float* movek  = (const float*)d_in[2];
    const float* movev  = (const float*)d_in[3];
    const float* wq     = (const float*)d_in[4];
    const float* bq     = (const float*)d_in[5];
    const float* wk     = (const float*)d_in[6];
    const float* bk     = (const float*)d_in[7];
    const float* wv     = (const float*)d_in[8];
    const float* bv     = (const float*)d_in[9];
    const float* lnq_g  = (const float*)d_in[10];
    const float* lnq_b  = (const float*)d_in[11];
    const float* lnk_g  = (const float*)d_in[12];
    const float* lnk_b  = (const float*)d_in[13];
    const float* lnv_g  = (const float*)d_in[14];
    const float* lnv_b  = (const float*)d_in[15];
    const float* lnc_g  = (const float*)d_in[16];
    const float* lnc_b  = (const float*)d_in[17];
    const float* rq_m1  = (const float*)d_in[18];
    const float* rq_a   = (const float*)d_in[19];
    const float* rq_m2  = (const float*)d_in[20];
    const float* rk_m1  = (const float*)d_in[21];
    const float* rk_a   = (const float*)d_in[22];
    const float* rk_m2  = (const float*)d_in[23];
    const float* rv_m1  = (const float*)d_in[24];
    const float* rv_a   = (const float*)d_in[25];
    const float* rv_m2  = (const float*)d_in[26];
    const float* rc_m1  = (const float*)d_in[27];
    const float* rc_a   = (const float*)d_in[28];
    const float* rc_m2  = (const float*)d_in[29];
    const float* moveq2 = (const float*)d_in[30];
    const float* movek2 = (const float*)d_in[31];
    const float* movev2 = (const float*)d_in[32];
    const float* clip   = (const float*)d_in[33];

    size_t off = 0;
    auto carve = [&](size_t bytes) -> void* {
        void* p = (void*)((char*)d_ws + off);
        off = (off + bytes + 255) & ~(size_t)255;
        return p;
    };
    const size_t WBYTES = (size_t)HH * HH;
    const size_t XBYTES = (size_t)MTOK * HH;
    const size_t FBYTES = (size_t)MTOK * HH * 4;

    signed char* wbq = (signed char*)carve(WBYTES);
    signed char* wbk = (signed char*)carve(WBYTES);
    signed char* wbv = (signed char*)carve(WBYTES);
    float* sclq = (float*)carve(HH * 4);
    float* sclk = (float*)carve(HH * 4);
    float* sclv = (float*)carve(HH * 4);
    signed char* xbq = (signed char*)carve(XBYTES);
    signed char* xbk = (signed char*)carve(XBYTES);
    signed char* xbv = (signed char*)carve(XBYTES);
    signed char* qb2 = (signed char*)carve(XBYTES);
    signed char* kb2 = (signed char*)carve(XBYTES);
    signed char* vb2 = (signed char*)carve(XBYTES);
    float* msum = (float*)carve(FBYTES);
    float* ctx  = (float*)carve(FBYTES);

    float* out = (float*)d_out;

    prep_w_kernel<<<HH, 256, 0, stream>>>(wq, wbq, sclq);
    prep_w_kernel<<<HH, 256, 0, stream>>>(wk, wbk, sclk);
    prep_w_kernel<<<HH, 256, 0, stream>>>(wv, wbv, sclv);

    const int total = MTOK * HH;
    binarize_x_kernel<<<(total + 255) / 256, 256, 0, stream>>>(
        x, moveq, movek, movev, xbq, xbk, xbv, total);

    gemm_ln_kernel<<<MT16, 256, 0, stream>>>(xbq, wbq, sclq, bq, lnq_g, lnq_b, x,
                                             rq_m1, rq_a, rq_m2, moveq2, msum, qb2, 0);
    gemm_ln_kernel<<<MT16, 256, 0, stream>>>(xbk, wbk, sclk, bk, lnk_g, lnk_b, x,
                                             rk_m1, rk_a, rk_m2, movek2, msum, kb2, 1);
    gemm_ln_kernel<<<MT16, 256, 0, stream>>>(xbv, wbv, sclv, bv, lnv_g, lnv_b, x,
                                             rv_m1, rv_a, rv_m2, movev2, msum, vb2, 1);

    attn_kernel<<<dim3(13, NHH, BB), 32, 0, stream>>>(qb2, kb2, vb2, clip, ctx);

    final_ln_kernel<<<MTOK, 256, 0, stream>>>(ctx, msum, lnc_g, lnc_b,
                                              rc_m1, rc_a, rc_m2, out);
}
